// JEPA_base_23390391894411
// MI455X (gfx1250) — compile-verified, matrix operand order checked by reading, not run on hardware
//
#include <hip/hip_runtime.h>
#include <math.h>

typedef __attribute__((ext_vector_type(16))) _Float16 v16h;
typedef __attribute__((ext_vector_type(8)))  _Float16 v8h;
typedef __attribute__((ext_vector_type(8)))  float    v8f;

#define TILE_M 256   // 8 waves x 32 rows
#define TILE_N 64
#define LDS_RS 40    // f16 row stride of transposed B tile (80B: 16B-aligned, conflict-light)

// ---------------------------------------------------------------------------
// WMMA GEMM: C[Mr,N] = act( A[Mr,K] @ B[K,N] + addsrc + bias )
// block = 256 threads (8 waves), block tile 256x64, wave tile 32x64.
// B k-chunks double-buffered in LDS transposed [n][k] as f16. All four B
// fragments are loaded into distinct registers before the 8 WMMAs so the
// ds_load_b128s pipeline instead of serializing on s_wait_dscnt per pair.
// ---------------------------------------------------------------------------
__global__ __launch_bounds__(256) void gemm_wmma_f16(
    const float* __restrict__ A, const float* __restrict__ Bm,
    float* __restrict__ C, const float* __restrict__ addsrc,
    const float* __restrict__ bias, int Mr, int N, int K, int act)
{
    __shared__ _Float16 lbT[2][64 * LDS_RS];   // (n, k) at n*LDS_RS + k
    const int tid  = threadIdx.x;
    const int lane = tid & 31;
    const int wave = tid >> 5;
    const int m0   = blockIdx.x * TILE_M + wave * 32;
    const int n0   = blockIdx.y * TILE_N;
    const int rowA0 = m0 + (lane & 15);
    const int laneHalf = (lane & 16) ? 8 : 0;   // K offset for upper lane group
    const int sn = tid & 63;                    // staging column (n within tile)
    const int sk = (tid >> 6) * 8;              // staging k base
    const int sidx = sn * LDS_RS + sk;

    v8f acc[2][4] = {};

    // running pointers (strength-reduced; advance by 32*N / 32 per chunk)
    const float* bstage = Bm + (size_t)sk * N + n0 + sn;
    const float* aptr0  = A + (size_t)rowA0 * K + laneHalf;
    const float* aptr1  = aptr0 + (size_t)16 * K;

    // prologue: stage first B chunk into buffer 0
    {
        v8h pk;
        #pragma unroll
        for (int i = 0; i < 8; i++) pk[i] = (_Float16)bstage[(size_t)i * N];
        *reinterpret_cast<v8h*>(&lbT[0][sidx]) = pk;  // ds_store_b128
    }
    __syncthreads();

    int buf = 0;
    for (int k0 = 0; k0 < K; k0 += 32) {
        // --- stage next B chunk into the other buffer (overlaps WMMA below) ---
        if (k0 + 32 < K) {
            const float* bp = bstage + (size_t)32 * N;
            v8h pk;
            #pragma unroll
            for (int i = 0; i < 8; i++) pk[i] = (_Float16)bp[(size_t)i * N];
            *reinterpret_cast<v8h*>(&lbT[buf ^ 1][sidx]) = pk;
            __builtin_prefetch(bp + (size_t)32 * N, 0, 1);
        }
        bstage += (size_t)32 * N;

        // --- two A fragments (rows rowA0, rowA0+16), ISA K-interleave layout ---
        v16h af[2];
        {
            v8f alo = *reinterpret_cast<const v8f*>(aptr0);
            v8f ahi = *reinterpret_cast<const v8f*>(aptr0 + 16);
            af[0] = __builtin_shufflevector(__builtin_convertvector(alo, v8h),
                                            __builtin_convertvector(ahi, v8h),
                                            0, 1, 2, 3, 4, 5, 6, 7,
                                            8, 9, 10, 11, 12, 13, 14, 15);
            v8f blo2 = *reinterpret_cast<const v8f*>(aptr1);
            v8f bhi2 = *reinterpret_cast<const v8f*>(aptr1 + 16);
            af[1] = __builtin_shufflevector(__builtin_convertvector(blo2, v8h),
                                            __builtin_convertvector(bhi2, v8h),
                                            0, 1, 2, 3, 4, 5, 6, 7,
                                            8, 9, 10, 11, 12, 13, 14, 15);
        }
        aptr0 += 32;
        aptr1 += 32;

        // --- load all 4 B fragments first (8 ds_load_b128, pipelined) ---
        const _Float16* lb = &lbT[buf][0];
        v16h bf[4];
        #pragma unroll
        for (int t = 0; t < 4; t++) {
            const int coln = t * 16 + (lane & 15);
            v8h blo = *reinterpret_cast<const v8h*>(&lb[coln * LDS_RS + laneHalf]);
            v8h bhi = *reinterpret_cast<const v8h*>(&lb[coln * LDS_RS + 16 + laneHalf]);
            bf[t] = __builtin_shufflevector(blo, bhi,
                        0, 1, 2, 3, 4, 5, 6, 7, 8, 9, 10, 11, 12, 13, 14, 15);
        }

        // --- 4 B tiles x 2 row blocks = 8 WMMAs per k-chunk ---
        #pragma unroll
        for (int t = 0; t < 4; t++) {
            acc[0][t] = __builtin_amdgcn_wmma_f32_16x16x32_f16(
                false, af[0], false, bf[t], (short)0, acc[0][t], false, false);
            acc[1][t] = __builtin_amdgcn_wmma_f32_16x16x32_f16(
                false, af[1], false, bf[t], (short)0, acc[1][t], false, false);
        }
        __syncthreads();   // next-buffer staging visible; safe to overwrite 'buf' next iter
        buf ^= 1;
    }

    // --- epilogue: residual add, bias, GELU, store (C/D ISA layout) ---
    #pragma unroll
    for (int r = 0; r < 2; r++) {
        #pragma unroll
        for (int t = 0; t < 4; t++) {
            const int col = n0 + t * 16 + (lane & 15);
            #pragma unroll
            for (int v = 0; v < 8; v++) {
                const int row = m0 + r * 16 + ((lane & 16) ? v + 8 : v);
                float x = acc[r][t][v];
                if (addsrc) x += addsrc[(size_t)row * N + col];
                if (bias)   x += bias[col];
                if (act) {
                    float u = x;
                    x = 0.5f * u * (1.0f + tanhf(0.7978845608f * (u + 0.044715f * u * u * u)));
                }
                C[(size_t)row * N + col] = x;
            }
        }
    }
}

// ---------------------------------------------------------------------------
// LayerNorm over D=256 (blockDim.x == 256). Optional affine, optional
// gather of the second half of a [B, 2*seqL, D] tensor.
// ---------------------------------------------------------------------------
__global__ void layernorm_kernel(const float* __restrict__ in, float* __restrict__ out,
                                 const float* __restrict__ g, const float* __restrict__ b2,
                                 int seqL, int halfsel)
{
    const int r = blockIdx.x;
    const int d = threadIdx.x;
    const int D = blockDim.x;
    size_t irow = halfsel ? ((size_t)(r / seqL) * (2 * seqL) + seqL + (r % seqL))
                          : (size_t)r;
    __shared__ float red[256];
    float x = in[irow * D + d];
    red[d] = x; __syncthreads();
    for (int s = 128; s > 0; s >>= 1) { if (d < s) red[d] += red[d + s]; __syncthreads(); }
    float mu = red[0] / (float)D;
    __syncthreads();
    float c = x - mu;
    red[d] = c * c; __syncthreads();
    for (int s = 128; s > 0; s >>= 1) { if (d < s) red[d] += red[d + s]; __syncthreads(); }
    float var = red[0] / (float)D;
    float y = c * rsqrtf(var + 1e-5f);
    if (g) y = y * g[d] + b2[d];
    out[(size_t)r * D + d] = y;
}

// ---------------------------------------------------------------------------
// Fused attention: one block per (b,h,q). hd fixed at 32, block = 128.
// Q/K/V/O layout: [B*S, D] with column h*32+d. Optional key padding mask.
// ---------------------------------------------------------------------------
__global__ __launch_bounds__(128) void attention_kernel(
    const float* __restrict__ Q, const float* __restrict__ Kt,
    const float* __restrict__ V, float* __restrict__ O,
    const int* __restrict__ np, int B, int H, int S)
{
    const int HD = 32;
    const int D  = H * HD;
    const int q  = blockIdx.x % S;
    const int h  = (blockIdx.x / S) % H;
    const int b  = blockIdx.x / (S * H);
    const int tid = threadIdx.x;

    __shared__ float sc[512];
    __shared__ float qv[32];
    __shared__ float red[128];
    __shared__ float oacc[4 * 32];

    if (tid < HD) qv[tid] = Q[((size_t)(b * S + q)) * D + h * HD + tid];
    __syncthreads();

    const int npv = np ? np[b] : S;
    const float scale = rsqrtf((float)HD);
    for (int k = tid; k < S; k += 128) {
        const float* kp = &Kt[((size_t)(b * S + k)) * D + h * HD];
        float s = 0.f;
        #pragma unroll
        for (int d2 = 0; d2 < 32; d2++) s += qv[d2] * kp[d2];
        s *= scale;
        if (k >= npv) s = -1e9f;
        sc[k] = s;
    }
    __syncthreads();

    float lm = -1e30f;
    for (int k = tid; k < S; k += 128) lm = fmaxf(lm, sc[k]);
    red[tid] = lm; __syncthreads();
    for (int s2 = 64; s2 > 0; s2 >>= 1) {
        if (tid < s2) red[tid] = fmaxf(red[tid], red[tid + s2]);
        __syncthreads();
    }
    float mx = red[0]; __syncthreads();

    float ls = 0.f;
    for (int k = tid; k < S; k += 128) { float e = __expf(sc[k] - mx); sc[k] = e; ls += e; }
    red[tid] = ls; __syncthreads();
    for (int s2 = 64; s2 > 0; s2 >>= 1) {
        if (tid < s2) red[tid] += red[tid + s2];
        __syncthreads();
    }
    float denom = red[0]; __syncthreads();

    const int d2 = tid & 31;
    const int part = tid >> 5;
    float a = 0.f;
    for (int k = part; k < S; k += 4)
        a += sc[k] * V[((size_t)(b * S + k)) * D + h * HD + d2];
    oacc[part * 32 + d2] = a; __syncthreads();
    if (part == 0) {
        float o = (oacc[d2] + oacc[32 + d2] + oacc[64 + d2] + oacc[96 + d2]) / denom;
        O[((size_t)(b * S + q)) * D + h * HD + d2] = o;
    }
}

// ---------------------------------------------------------------------------
// Hash RNG (deterministic, structure-matching stand-in for jax.random).
// ---------------------------------------------------------------------------
__device__ inline unsigned hashu(unsigned x) {
    x ^= x >> 16; x *= 0x7feb352du; x ^= x >> 15; x *= 0x846ca68bu; x ^= x >> 16;
    return x;
}
__device__ inline float u01(unsigned a, unsigned b) {
    unsigned h = hashu(a ^ (hashu(b) + 0x9e3779b9u + (a << 6) + (a >> 2)));
    return (float)(h >> 8) * (1.0f / 16777216.0f);
}

// all_masks[m,b,l]: successive span OR rank-based random subset, minus padding.
__global__ void masks_kernel(const int* __restrict__ np, int* __restrict__ all_mask,
                             int B, int L)
{
    const int m = blockIdx.x / B, b = blockIdx.x % B;
    const int l = threadIdx.x;
    const int npv = np[b];
    const bool pad = l >= npv;
    const float RATS[3] = {0.15f, 0.25f, 0.35f};
    const float r = RATS[hashu(0xBEEFu) % 3u];
    const int num_mask = (int)floorf((float)npv * r);
    const bool succ = u01(1001u + (unsigned)m, (unsigned)b) < 0.5f;
    const int start = (int)floorf(u01(2002u + (unsigned)m, (unsigned)b) *
                                  ((float)npv - (float)num_mask + 1.0f));
    const float sc = pad ? 2.0f : u01(3003u + (unsigned)(m * B + b), (unsigned)l);
    __shared__ float ssc[256];
    ssc[l] = sc; __syncthreads();
    int rank = 0;
    for (int j = 0; j < L; j++) {
        float vv = ssc[j];
        rank += (vv < sc) || (vv == sc && j < l);
    }
    const bool rm = rank < num_mask;
    const bool sm = (l >= start) && (l < start + num_mask);
    all_mask[(m * B + b) * L + l] = ((succ ? sm : rm) && !pad) ? 1 : 0;
}

// sample_mask[b,l]: rank of random score < floor(nonpad * clip(u,0.6,1)).
__global__ void sample_kernel(const int* __restrict__ np, int* __restrict__ sample, int L)
{
    const int b = blockIdx.x;
    const int l = threadIdx.x;
    const int npv = np[b];
    const bool pad = l >= npv;
    float ratio = u01(4004u, (unsigned)b);
    ratio = fminf(fmaxf(ratio, 0.6f), 1.0f);
    const int num_samp = (int)floorf((float)npv * ratio);
    const float sc = pad ? 2.0f : u01(5005u + (unsigned)b, (unsigned)l);
    __shared__ float ssc[256];
    ssc[l] = sc; __syncthreads();
    int rank = 0;
    for (int j = 0; j < L; j++) {
        float vv = ssc[j];
        rank += (vv < sc) || (vv == sc && j < l);
    }
    sample[b * L + l] = (rank < num_samp) ? 1 : 0;
}

__global__ void valid_kernel(const int* __restrict__ all_mask, const int* __restrict__ sample,
                             int* __restrict__ valid, int n, int BL)
{
    int idx = blockIdx.x * blockDim.x + threadIdx.x;
    if (idx < n) valid[idx] = (sample[idx % BL] && !all_mask[idx]) ? 1 : 0;
}

__global__ void adjmean_kernel(const float* __restrict__ adj, float* __restrict__ out, int n, int D)
{
    int idx = blockIdx.x * blockDim.x + threadIdx.x;
    if (idx >= n) return;
    int row = idx / D, d = idx % D;
    float s = 0.f;
    #pragma unroll
    for (int j = 0; j < 9; j++) s += adj[((size_t)row * 9 + j) * D + d];
    out[idx] = s * (1.0f / 9.0f);
}

__global__ void add_traj_kernel(float* __restrict__ h, const float* __restrict__ traj,
                                const float* __restrict__ w, int n, int D)
{
    int idx = blockIdx.x * blockDim.x + threadIdx.x;
    if (idx >= n) return;
    int row = idx / D, d = idx % D;
    h[idx] += traj[row * 2] * w[d] + traj[row * 2 + 1] * w[D + d];
}

__global__ void mask_rows_kernel(const float* __restrict__ cell, const float* __restrict__ adjm,
                                 const int* __restrict__ validm, float* __restrict__ mA,
                                 float* __restrict__ mB, int n, int D)
{
    int idx = blockIdx.x * blockDim.x + threadIdx.x;
    if (idx >= n) return;
    float w = (float)validm[idx / D];
    mA[idx] = cell[idx] * w;
    mB[idx] = adjm[idx] * w;
}

__global__ void target_out_kernel(const float* __restrict__ xemb, const int* __restrict__ all_mask,
                                  float* __restrict__ out, int n, int chunk, int D)
{
    int idx = blockIdx.x * blockDim.x + threadIdx.x;
    if (idx >= n) return;
    out[idx] = xemb[idx % chunk] * (float)all_mask[idx / D];
}

// h_pred[b, 0:L] = context encoder output; h_pred[b, L:2L] = latent + pe*mask
__global__ void build_pred_kernel(const float* __restrict__ ctxout, const float* __restrict__ latent,
                                  const float* __restrict__ pe, const int* __restrict__ amask,
                                  float* __restrict__ hp, int L, int D, int n)
{
    int idx = blockIdx.x * blockDim.x + threadIdx.x;
    if (idx >= n) return;
    int d = idx % D;
    int s = (idx / D) % (2 * L);
    int b = idx / (D * 2 * L);
    if (s < L) {
        hp[idx] = ctxout[((size_t)b * L + s) * D + d];
    } else {
        int l = s - L;
        hp[idx] = latent[d] + pe[(size_t)l * D + d] * (float)amask[b * L + l];
    }
}

// ---------------------------------------------------------------------------
// Host orchestration
// ---------------------------------------------------------------------------
extern "C" void kernel_launch(void* const* d_in, const int* in_sizes, int n_in,
                              void* d_out, int out_size, void* d_ws, size_t ws_size,
                              hipStream_t stream)
{
    (void)in_sizes; (void)n_in; (void)out_size; (void)ws_size;
    const float* cell      = (const float*)d_in[0];
    const float* traj      = (const float*)d_in[1];
    const float* adj       = (const float*)d_in[2];
    const int*   npts      = (const int*)d_in[3];
    const float* tgt_w_adj = (const float*)d_in[4];
    const float* tgt_w_trj = (const float*)d_in[5];
    const float* tgt_attn  = (const float*)d_in[6];
    const float* tgt_ffn1  = (const float*)d_in[7];
    const float* tgt_ffn2  = (const float*)d_in[8];
    const float* ctx_w_adj = (const float*)d_in[9];
    const float* ctx_attn  = (const float*)d_in[10];
    const float* ctx_ffn1  = (const float*)d_in[11];
    const float* ctx_ffn2  = (const float*)d_in[12];
    const float* pe        = (const float*)d_in[13];
    const float* latent    = (const float*)d_in[14];
    const float* pred_attn = (const float*)d_in[15];
    const float* pred_ffn1 = (const float*)d_in[16];
    const float* pred_ffn2 = (const float*)d_in[17];
    const float* norm_g    = (const float*)d_in[18];
    const float* norm_b    = (const float*)d_in[19];
    const float* proj_w    = (const float*)d_in[20];
    const float* proj_b    = (const float*)d_in[21];
    float* out = (float*)d_out;

    constexpr int B = 8, L = 256, D = 256, NL = 4, PD = 2, H = 8, Mseg = 4;
    const int rowsE = B * L;       // 2048
    const int rowsP = B * 2 * L;   // 4096

    // ---- workspace carve-out ----
    float* p = (float*)d_ws;
    auto alloc = [&](size_t nf) { float* r = p; p += nf; return r; };
    float* adjm   = alloc((size_t)rowsE * D);
    float* xemb   = alloc((size_t)rowsE * D);
    float* ctxout = alloc((size_t)rowsE * D);
    float* mA     = alloc((size_t)rowsE * D);
    float* mB     = alloc((size_t)rowsE * D);
    float* hA     = alloc((size_t)rowsP * D);
    float* hB     = alloc((size_t)rowsP * D);
    float* lnbuf  = alloc((size_t)rowsP * D);
    float* qb     = alloc((size_t)rowsP * D);
    float* kb     = alloc((size_t)rowsP * D);
    float* vb     = alloc((size_t)rowsP * D);
    float* attb   = alloc((size_t)rowsP * D);
    float* ffnb   = qb;  // FFN intermediate aliases dead q/k/v/att block (exactly 4x)
    int* all_mask = (int*)p;
    int* sample   = all_mask + Mseg * B * L;
    int* valid    = sample + B * L;

    auto ew = [&](int n) { return dim3((n + 255) / 256); };

    auto gemm = [&](const float* A, const float* W, float* C, int Mr, int N, int K,
                    const float* add, const float* bias, int act) {
        dim3 g(Mr / TILE_M, N / TILE_N);
        gemm_wmma_f16<<<g, 256, 0, stream>>>(A, W, C, add, bias, Mr, N, K, act);
    };
    auto ln = [&](const float* in, float* o, const float* g, const float* b2,
                  int rows, int seqL, int halfsel) {
        layernorm_kernel<<<rows, 256, 0, stream>>>(in, o, g, b2, seqL, halfsel);
    };

    // pre-LN transformer stack; h in/out in `h`, `h2` is ping buffer
    auto run_layers = [&](float* h, float* h2, int rows, int S, const float* attnW,
                          const float* f1, const float* f2, int nl, const int* npp) {
        for (int l = 0; l < nl; l++) {
            const float* Wq = attnW + ((size_t)l * 4 + 0) * D * D;
            const float* Wk = attnW + ((size_t)l * 4 + 1) * D * D;
            const float* Wv = attnW + ((size_t)l * 4 + 2) * D * D;
            const float* Wo = attnW + ((size_t)l * 4 + 3) * D * D;
            ln(h, lnbuf, nullptr, nullptr, rows, S, 0);
            gemm(lnbuf, Wq, qb, rows, D, D, nullptr, nullptr, 0);
            gemm(lnbuf, Wk, kb, rows, D, D, nullptr, nullptr, 0);
            gemm(lnbuf, Wv, vb, rows, D, D, nullptr, nullptr, 0);
            attention_kernel<<<B * H * S, 128, 0, stream>>>(qb, kb, vb, attb, npp, B, H, S);
            gemm(attb, Wo, h2, rows, D, D, /*add*/ h, nullptr, 0);
            ln(h2, lnbuf, nullptr, nullptr, rows, S, 0);
            gemm(lnbuf, f1 + (size_t)l * D * 4 * D, ffnb, rows, 4 * D, D, nullptr, nullptr, 1);
            gemm(ffnb, f2 + (size_t)l * 4 * D * D, h, rows, D, 4 * D, /*add*/ h2, nullptr, 0);
        }
    };

    // ---- shared preprocessing ----
    adjmean_kernel<<<ew(rowsE * D), 256, 0, stream>>>(adj, adjm, rowsE * D, D);

    // ---- target encoder ----
    gemm(adjm, tgt_w_adj, hA, rowsE, D, D, /*add*/ cell, nullptr, 0);
    add_traj_kernel<<<ew(rowsE * D), 256, 0, stream>>>(hA, traj, tgt_w_trj, rowsE * D, D);
    run_layers(hA, hB, rowsE, L, tgt_attn, tgt_ffn1, tgt_ffn2, NL, npts);
    ln(hA, xemb, nullptr, nullptr, rowsE, L, 0);

    // ---- masks ----
    masks_kernel<<<Mseg * B, 256, 0, stream>>>(npts, all_mask, B, L);
    sample_kernel<<<B, 256, 0, stream>>>(npts, sample, L);
    valid_kernel<<<ew(Mseg * B * L), 256, 0, stream>>>(all_mask, sample, valid,
                                                       Mseg * B * L, B * L);
    target_out_kernel<<<ew(Mseg * rowsE * D), 256, 0, stream>>>(
        xemb, all_mask, out + (size_t)Mseg * rowsE * D, Mseg * rowsE * D, rowsE * D, D);

    // ---- per-mask-segment context encoder + predictor ----
    for (int m = 0; m < Mseg; m++) {
        mask_rows_kernel<<<ew(rowsE * D), 256, 0, stream>>>(
            cell, adjm, valid + m * B * L, mA, mB, rowsE * D, D);
        gemm(mB, ctx_w_adj, hA, rowsE, D, D, /*add*/ mA, nullptr, 0);
        run_layers(hA, hB, rowsE, L, ctx_attn, ctx_ffn1, ctx_ffn2, NL, npts);
        ln(hA, ctxout, nullptr, nullptr, rowsE, L, 0);

        build_pred_kernel<<<ew(rowsP * D), 256, 0, stream>>>(
            ctxout, latent, pe, all_mask + m * B * L, hA, L, D, rowsP * D);
        run_layers(hA, hB, rowsP, 2 * L, pred_attn, pred_ffn1, pred_ffn2, PD, nullptr);

        // LN(h[:, L:]) * g + b  ->  @ proj_w + proj_b
        ln(hA, lnbuf, norm_g, norm_b, rowsE, L, /*halfsel*/ 1);
        gemm(lnbuf, proj_w, out + (size_t)m * rowsE * D, rowsE, D, D, nullptr, proj_b, 0);
    }
}
